// DynamicHierarchyAttention_28269474742695
// MI455X (gfx1250) — compile-verified, matrix operand order checked by reading, not run on hardware
//
#include <hip/hip_runtime.h>

typedef __attribute__((ext_vector_type(16))) __bf16 v16bf;
typedef __attribute__((ext_vector_type(8)))  float  v8f;

#define HS 150
#define G3 450          // 3*HS
#define NT_H 29         // ceil(450/16) n-tiles of the gate projection
#define KP_E 320        // padded input dim (300 -> 320)
#define KP_H 160        // padded hidden (150 -> 160)
#define LW 64
#define NB 64
#define NS 20
#define NSEQ_S 1344     // 64 target + 64*20 ctx sequences
#define EDIM 300

__device__ __forceinline__ unsigned short f2bf(float f) {
  unsigned u = __float_as_uint(f);
  unsigned r = (u + 0x7FFFu + ((u >> 16) & 1u)) >> 16;   // RNE
  return (unsigned short)r;
}
__device__ __forceinline__ float bf2f(unsigned short h) {
  return __uint_as_float(((unsigned)h) << 16);
}

union BF2 { unsigned u; __bf16 h[2]; };

// Build a 16x32 bf16 WMMA fragment (A or B) from a row-major (row, K) buffer.
// ISA 7.12.2 16-bit layout: lane m = lane&15; lanes 0-15 hold K 0-7,16-23;
// lanes 16-31 hold K 8-15,24-31; each VGPR = 2 consecutive K (pair-loaded as b32).
__device__ __forceinline__ v16bf ldfrag(const unsigned short* base, int row,
                                        int ld, int kk, int lane) {
  const unsigned* p = (const unsigned*)(base + (size_t)row * ld + kk * 32);
  int g = (lane >> 4) & 1;
  v16bf f;
#pragma unroll
  for (int v = 0; v < 8; ++v) {
    int kp = (v < 4) ? (g * 4 + v) : (8 + g * 4 + (v - 4)); // pair index k/2
    BF2 c; c.u = p[kp];
    f[2 * v]     = c.h[0];
    f[2 * v + 1] = c.h[1];
  }
  return f;
}

// f32 (R,C) row-major -> bf16 (Rp,Cp) row-major, zero padded.
__global__ void pad_convert(const float* __restrict__ src, unsigned short* __restrict__ dst,
                            int R, int C, int Rp, int Cp) {
  int total = Rp * Cp;
  for (int i = blockIdx.x * blockDim.x + threadIdx.x; i < total; i += gridDim.x * blockDim.x) {
    int r = i / Cp, c = i % Cp;
    float v = (r < R && c < C) ? src[(size_t)r * C + c] : 0.f;
    dst[i] = f2bf(v);
  }
}

// Fused bidirectional-GRU direction kernel.
// mode 0: inputs are embedding gathers of target/ctx token ids (E=300).
// mode 1: inputs are a dense f32 (nseq, T, 300) buffer.
// Each workgroup owns 16 sequences; per step: build bf16 X tile, WMMA
// x@Wih^T (K=320) and h@Whh^T (K=160) per 16-col tile, then f32 gates.
__global__ __launch_bounds__(256)
void gru_fused(const float* __restrict__ word_emb, const int* __restrict__ tids,
               const int* __restrict__ cids, const int* __restrict__ tlens,
               const int* __restrict__ clens, const float* __restrict__ Xdense,
               const unsigned short* __restrict__ Wih_f, const unsigned short* __restrict__ Wih_b,
               const unsigned short* __restrict__ Whh_f, const unsigned short* __restrict__ Whh_b,
               const float* __restrict__ bih_f, const float* __restrict__ bih_b,
               const float* __restrict__ bhh_f, const float* __restrict__ bhh_b,
               unsigned short* __restrict__ Obf, int T, int mode) {
  __shared__ __align__(16) float          sh_h[16][KP_H];     // 10240 B
  __shared__ __align__(16) unsigned short sh_hbf[16][KP_H];   //  5120 B
  __shared__ __align__(16) unsigned short sh_xbf[16][KP_E];   // 10240 B
  __shared__ __align__(16) float          sh_g[16][452];      // 28928 B (xr+hr | xz+hz | xn)
  __shared__ __align__(16) float          sh_hn[16][152];     //  9728 B (hn)
  __shared__ int sh_len[16];

  const int tid  = threadIdx.x;
  const int lane = tid & 31;
  const int wave = tid >> 5;
  const int dir  = blockIdx.y;
  const int seqbase = blockIdx.x * 16;

  const unsigned short* Wih = dir ? Wih_b : Wih_f;
  const unsigned short* Whh = dir ? Whh_b : Whh_f;
  const float* bih = dir ? bih_b : bih_f;
  const float* bhh = dir ? bhh_b : bhh_f;

  for (int i = tid; i < 16 * KP_H; i += 256) {
    (&sh_h[0][0])[i] = 0.f;
    (&sh_hbf[0][0])[i] = 0;
  }
  if (tid < 16) {
    int seq = seqbase + tid;
    sh_len[tid] = (mode == 0) ? ((seq < NB) ? tlens[seq] : clens[seq - NB]) : T;
  }
  __syncthreads();

  for (int tt = 0; tt < T; ++tt) {
    const int t = dir ? (T - 1 - tt) : tt;

    // ---- build bf16 input tile (16 x 320, zero padded) ----
    for (int i = tid; i < 16 * KP_E; i += 256) {
      int sq = i / KP_E, c = i % KP_E;
      int seq = seqbase + sq;
      float v = 0.f;
      if (c < EDIM) {
        if (mode == 0) {
          int id = (seq < NB) ? tids[seq * LW + t] : cids[(seq - NB) * LW + t];
          v = word_emb[(size_t)id * EDIM + c];
        } else {
          v = Xdense[((size_t)seq * T + t) * EDIM + c];
        }
      }
      sh_xbf[sq][c] = f2bf(v);
    }
    __syncthreads();

    // ---- WMMA: xp = X @ Wih^T, hp = h @ Whh^T per 16-col tile ----
    for (int nt = wave; nt < NT_H; nt += 8) {
      v8f xacc = {};
      v8f hacc = {};
      const int arow = lane & 15;
      const int brow = nt * 16 + (lane & 15);
#pragma unroll
      for (int kk = 0; kk < KP_E / 32; ++kk) {
        v16bf a = ldfrag(&sh_xbf[0][0], arow, KP_E, kk, lane);
        v16bf b = ldfrag(Wih, brow, KP_E, kk, lane);
        xacc = __builtin_amdgcn_wmma_f32_16x16x32_bf16(false, a, false, b,
                                                       (short)0, xacc, false, false);
      }
#pragma unroll
      for (int kk = 0; kk < KP_H / 32; ++kk) {
        v16bf a = ldfrag(&sh_hbf[0][0], arow, KP_H, kk, lane);
        v16bf b = ldfrag(Whh, brow, KP_H, kk, lane);
        hacc = __builtin_amdgcn_wmma_f32_16x16x32_bf16(false, a, false, b,
                                                       (short)0, hacc, false, false);
      }
      const int col = nt * 16 + (lane & 15);
      const int g = (lane >> 4) & 1;
      if (col < G3) {
#pragma unroll
        for (int i = 0; i < 8; ++i) {
          int m = i + 8 * g;                 // C/D layout: VGPR i -> row i (+8 for hi half)
          if (col < 2 * HS) {
            sh_g[m][col] = xacc[i] + hacc[i];      // r,z pre-activations combined
          } else {
            sh_g[m][col] = xacc[i];                // xn
            sh_hn[m][col - 2 * HS] = hacc[i];      // hn
          }
        }
      }
    }
    __syncthreads();

    // ---- gates (f32) + state update ----
    for (int i = tid; i < 16 * HS; i += 256) {
      int sq = i / HS, j = i % HS;
      int seq = seqbase + sq;
      float r = 1.f / (1.f + expf(-(sh_g[sq][j] + bih[j] + bhh[j])));
      float z = 1.f / (1.f + expf(-(sh_g[sq][HS + j] + bih[HS + j] + bhh[HS + j])));
      float n = tanhf(sh_g[sq][2 * HS + j] + bih[2 * HS + j] +
                      r * (sh_hn[sq][j] + bhh[2 * HS + j]));
      float hprev = sh_h[sq][j];
      float hcand = (1.f - z) * n + z * hprev;
      float hout = (t < sh_len[sq]) ? hcand : hprev;
      sh_h[sq][j] = hout;
      sh_hbf[sq][j] = f2bf(hout);
      Obf[((size_t)seq * T + t) * KP_E + dir * HS + j] = f2bf(hout);
    }
    if (dir == 0) {  // zero K-pad cols 300..319 of the output (A-matrix use later)
      for (int i = tid; i < 16 * (KP_E - 2 * HS); i += 256) {
        int sq = i / (KP_E - 2 * HS), c = 2 * HS + i % (KP_E - 2 * HS);
        int seq = seqbase + sq;
        Obf[((size_t)seq * T + t) * KP_E + c] = 0;
      }
    }
    __syncthreads();
  }
}

// C(M,Nvalid) = tanh?( A(M,320)bf16 @ W(Npad,320)bf16^T + bias ), one wave per 16x16 tile.
__global__ __launch_bounds__(128)
void gemm_wmma(const unsigned short* __restrict__ A, const unsigned short* __restrict__ W,
               const float* __restrict__ bias, float* __restrict__ C,
               int Nvalid, int ldC, int act_tanh) {
  const int tid = threadIdx.x;
  const int lane = tid & 31;
  const int wave = tid >> 5;
  const int mt = blockIdx.x;
  const int nt = blockIdx.y * 4 + wave;
  if (nt * 16 >= Nvalid) return;
  const int arow = mt * 16 + (lane & 15);
  const int brow = nt * 16 + (lane & 15);
  v8f acc = {};
#pragma unroll
  for (int kk = 0; kk < KP_E / 32; ++kk) {
    __builtin_prefetch((const void*)(W + (size_t)brow * KP_E + (kk + 1) * 32), 0, 0);
    v16bf a = ldfrag(A, arow, KP_E, kk, lane);
    v16bf b = ldfrag(W, brow, KP_E, kk, lane);
    acc = __builtin_amdgcn_wmma_f32_16x16x32_bf16(false, a, false, b,
                                                  (short)0, acc, false, false);
  }
  const int col = nt * 16 + (lane & 15);
  if (col < Nvalid) {
    const int g = (lane >> 4) & 1;
    float bv = bias[col];
#pragma unroll
    for (int i = 0; i < 8; ++i) {
      int row = mt * 16 + i + 8 * g;
      float v = acc[i] + bv;
      if (act_tanh) v = tanhf(v);
      C[(size_t)row * ldC + col] = v;
    }
  }
}

// Sentence attention pooling: scores = P@s_ctx (masked) -> softmax over T -> sent = attn^T O.
__global__ __launch_bounds__(128)
void pool_attn(const float* __restrict__ P, const unsigned short* __restrict__ Obf,
               const float* __restrict__ sctx, const int* __restrict__ tlens,
               const int* __restrict__ clens, float* __restrict__ out,
               float* __restrict__ tgt_sent, unsigned short* __restrict__ tgt_bf,
               float* __restrict__ ctxX) {
  __shared__ float at[LW];
  const int seq = blockIdx.x;
  const int tid = threadIdx.x;
  const int len = (seq < NB) ? tlens[seq] : clens[seq - NB];
  if (tid < LW) {
    float d = 0.f;
    const float* pr = P + ((size_t)seq * LW + tid) * EDIM;
    for (int c = 0; c < EDIM; ++c) d += pr[c] * sctx[c];
    at[tid] = (tid < len) ? d : -1e9f;
  }
  __syncthreads();
  if (tid == 0) {
    float mx = -3.4e38f;
    for (int t2 = 0; t2 < LW; ++t2) mx = fmaxf(mx, at[t2]);
    float s = 0.f;
    for (int t2 = 0; t2 < LW; ++t2) { at[t2] = expf(at[t2] - mx); s += at[t2]; }
    float inv = 1.f / s;
    for (int t2 = 0; t2 < LW; ++t2) at[t2] *= inv;
  }
  __syncthreads();
  if (seq < NB && tid < LW) out[2 * NB + seq * LW + tid] = at[tid];   // sent_attn
  for (int d = tid; d < EDIM; d += 128) {
    float s = 0.f;
    for (int t2 = 0; t2 < LW; ++t2)
      s += at[t2] * bf2f(Obf[((size_t)seq * LW + t2) * KP_E + d]);
    if (seq < NB) { tgt_sent[seq * EDIM + d] = s; tgt_bf[seq * KP_E + d] = f2bf(s); }
    else          { ctxX[(size_t)(seq - NB) * EDIM + d] = s; }
  }
  if (seq < NB) {
    for (int c = EDIM + tid; c < KP_E; c += 128) tgt_bf[seq * KP_E + c] = 0;
  }
}

// Message attention + logits.
__global__ __launch_bounds__(64)
void final_kernel(const float* __restrict__ tproj, const float* __restrict__ mproj,
                  const unsigned short* __restrict__ MObf, const float* __restrict__ tgt_sent,
                  const float* __restrict__ likW, const float* __restrict__ likb,
                  float* __restrict__ out) {
  __shared__ float at[NS];
  __shared__ float msg[EDIM];
  const int b = blockIdx.x, tid = threadIdx.x;
  if (tid < NS) {
    float d = 0.f;
    const float* mp = mproj + ((size_t)b * NS + tid) * EDIM;
    const float* tp = tproj + (size_t)b * EDIM;
    for (int c = 0; c < EDIM; ++c) d += mp[c] * tp[c];
    at[tid] = d;
  }
  __syncthreads();
  if (tid == 0) {
    float mx = -3.4e38f;
    for (int s = 0; s < NS; ++s) mx = fmaxf(mx, at[s]);
    float ss = 0.f;
    for (int s = 0; s < NS; ++s) { at[s] = expf(at[s] - mx); ss += at[s]; }
    float inv = 1.f / ss;
    for (int s = 0; s < NS; ++s) at[s] *= inv;
  }
  __syncthreads();
  if (tid < NS) out[2 * NB + NB * LW + b * NS + tid] = at[tid];       // msg_attn
  for (int d = tid; d < EDIM; d += 64) {
    float m = 0.f;
    for (int s = 0; s < NS; ++s) m += at[s] * bf2f(MObf[((size_t)b * NS + s) * KP_E + d]);
    msg[d] = m;
  }
  __syncthreads();
  if (tid < 2) {
    float l = likb[tid];
    const float* w = likW + tid * (2 * EDIM);
    for (int d = 0; d < EDIM; ++d)
      l += w[d] * msg[d] + w[EDIM + d] * tgt_sent[(size_t)b * EDIM + d];
    out[b * 2 + tid] = l;                                             // logits
  }
}

extern "C" void kernel_launch(void* const* d_in, const int* in_sizes, int n_in,
                              void* d_out, int out_size, void* d_ws, size_t ws_size,
                              hipStream_t stream) {
  (void)in_sizes; (void)n_in; (void)out_size; (void)ws_size;
  const int*   tids   = (const int*)d_in[0];
  const int*   tlens  = (const int*)d_in[1];
  const int*   cids   = (const int*)d_in[2];
  const int*   clens  = (const int*)d_in[3];
  const float* wemb   = (const float*)d_in[4];
  const float* sWih_f = (const float*)d_in[5];
  const float* sWhh_f = (const float*)d_in[6];
  const float* sbih_f = (const float*)d_in[7];
  const float* sbhh_f = (const float*)d_in[8];
  const float* sWih_b = (const float*)d_in[9];
  const float* sWhh_b = (const float*)d_in[10];
  const float* sbih_b = (const float*)d_in[11];
  const float* sbhh_b = (const float*)d_in[12];
  const float* sp_W   = (const float*)d_in[13];
  const float* sp_b   = (const float*)d_in[14];
  const float* s_ctx  = (const float*)d_in[15];
  const float* mWih_f = (const float*)d_in[16];
  const float* mWhh_f = (const float*)d_in[17];
  const float* mbih_f = (const float*)d_in[18];
  const float* mbhh_f = (const float*)d_in[19];
  const float* mWih_b = (const float*)d_in[20];
  const float* mWhh_b = (const float*)d_in[21];
  const float* mbih_b = (const float*)d_in[22];
  const float* mbhh_b = (const float*)d_in[23];
  const float* tgt_W  = (const float*)d_in[24];
  const float* tgt_b  = (const float*)d_in[25];
  const float* mp_W   = (const float*)d_in[26];
  const float* mp_b   = (const float*)d_in[27];
  const float* lik_W  = (const float*)d_in[28];
  const float* lik_b  = (const float*)d_in[29];
  float* out = (float*)d_out;

  // ---- workspace carving ----
  char* p = (char*)d_ws;
  auto take = [&](size_t bytes) -> void* {
    void* r = (void*)p;
    p += (bytes + 255) & ~(size_t)255;
    return r;
  };
  unsigned short* wihSf = (unsigned short*)take((size_t)464 * 320 * 2);
  unsigned short* wihSb = (unsigned short*)take((size_t)464 * 320 * 2);
  unsigned short* whhSf = (unsigned short*)take((size_t)464 * 160 * 2);
  unsigned short* whhSb = (unsigned short*)take((size_t)464 * 160 * 2);
  unsigned short* wihMf = (unsigned short*)take((size_t)464 * 320 * 2);
  unsigned short* wihMb = (unsigned short*)take((size_t)464 * 320 * 2);
  unsigned short* whhMf = (unsigned short*)take((size_t)464 * 160 * 2);
  unsigned short* whhMb = (unsigned short*)take((size_t)464 * 160 * 2);
  unsigned short* spWbf  = (unsigned short*)take((size_t)304 * 320 * 2);
  unsigned short* tgtWbf = (unsigned short*)take((size_t)304 * 320 * 2);
  unsigned short* mpWbf  = (unsigned short*)take((size_t)304 * 320 * 2);
  unsigned short* Obf    = (unsigned short*)take((size_t)NSEQ_S * LW * KP_E * 2);
  float*          Pproj  = (float*)take((size_t)NSEQ_S * LW * EDIM * 4);
  float*          tgtSent= (float*)take((size_t)NB * EDIM * 4);
  unsigned short* tgtBf  = (unsigned short*)take((size_t)NB * KP_E * 2);
  float*          ctxX   = (float*)take((size_t)NB * NS * EDIM * 4);
  unsigned short* MObf   = (unsigned short*)take((size_t)NB * NS * KP_E * 2);
  float*          tproj  = (float*)take((size_t)NB * EDIM * 4);
  float*          mproj  = (float*)take((size_t)NB * NS * EDIM * 4);

  // ---- weight conversions to padded bf16 ----
  auto cvt = [&](const float* s, unsigned short* d, int R, int C, int Rp, int Cp) {
    int total = Rp * Cp;
    pad_convert<<<(total + 255) / 256, 256, 0, stream>>>(s, d, R, C, Rp, Cp);
  };
  cvt(sWih_f, wihSf, G3, EDIM, 464, KP_E);
  cvt(sWih_b, wihSb, G3, EDIM, 464, KP_E);
  cvt(sWhh_f, whhSf, G3, HS, 464, KP_H);
  cvt(sWhh_b, whhSb, G3, HS, 464, KP_H);
  cvt(mWih_f, wihMf, G3, EDIM, 464, KP_E);
  cvt(mWih_b, wihMb, G3, EDIM, 464, KP_E);
  cvt(mWhh_f, whhMf, G3, HS, 464, KP_H);
  cvt(mWhh_b, whhMb, G3, HS, 464, KP_H);
  cvt(sp_W,  spWbf,  EDIM, EDIM, 304, KP_E);
  cvt(tgt_W, tgtWbf, EDIM, EDIM, 304, KP_E);
  cvt(mp_W,  mpWbf,  EDIM, EDIM, 304, KP_E);

  // ---- sentence-level bi-GRU (fused embed + x-proj + recurrence), both dirs ----
  gru_fused<<<dim3(NSEQ_S / 16, 2), 256, 0, stream>>>(
      wemb, tids, cids, tlens, clens, nullptr,
      wihSf, wihSb, whhSf, whhSb, sbih_f, sbih_b, sbhh_f, sbhh_b,
      Obf, LW, 0);

  // ---- proj = tanh(O @ sp_W^T + sp_b) ----
  gemm_wmma<<<dim3(NSEQ_S * LW / 16, 5), 128, 0, stream>>>(
      Obf, spWbf, sp_b, Pproj, EDIM, EDIM, 1);

  // ---- attention pooling over tokens ----
  pool_attn<<<NSEQ_S, 128, 0, stream>>>(Pproj, Obf, s_ctx, tlens, clens,
                                        out, tgtSent, tgtBf, ctxX);

  // ---- message-level bi-GRU over S=20 (dense-input mode) ----
  gru_fused<<<dim3(NB / 16, 2), 256, 0, stream>>>(
      nullptr, nullptr, nullptr, nullptr, nullptr, ctxX,
      wihMf, wihMb, whhMf, whhMb, mbih_f, mbih_b, mbhh_f, mbhh_b,
      MObf, NS, 1);

  // ---- tproj / mproj ----
  gemm_wmma<<<dim3(NB / 16, 5), 128, 0, stream>>>(
      tgtBf, tgtWbf, tgt_b, tproj, EDIM, EDIM, 1);
  gemm_wmma<<<dim3(NB * NS / 16, 5), 128, 0, stream>>>(
      MObf, mpWbf, mp_b, mproj, EDIM, EDIM, 1);

  // ---- message attention + logits ----
  final_kernel<<<NB, 64, 0, stream>>>(tproj, mproj, MObf, tgtSent, lik_W, lik_b, out);
}
